// Quantizer_29884382446082
// MI455X (gfx1250) — matrix operand order repaired and checked by reading, not compiled, not run on hardware
//
#include <hip/hip_runtime.h>
#include <hip/hip_bf16.h>

#define B_ROWS 131072
#define HL 32
#define NCODES 256
#define NLVL 3
#define LN_EPS 1e-5f
#define BETA 0.25f

typedef __attribute__((ext_vector_type(16))) __bf16 v16bf;
typedef __attribute__((ext_vector_type(8)))  __bf16 v8bf;
typedef __attribute__((ext_vector_type(8)))  float  v8f;

// ---------------- weight convert+transpose: f32 [K,N] -> bf16 [N,K] ----------------
__global__ void wconv_kernel(const float* __restrict__ src, __bf16* __restrict__ dst,
                             int K, int N) {
  int i = blockIdx.x * blockDim.x + threadIdx.x;
  if (i < K * N) {
    int k = i / N, n = i - k * N;
    dst[(size_t)n * K + k] = (__bf16)src[i];
  }
}

// ---------------- generic WMMA GEMM: C = act(A[M,K] @ Wt[N,K]^T + bias) ----------------
// 256 threads = 8 waves; block tile 128 x (16*NSUB); K-step 32.
// B fragments consumed in double-buffered groups of 4 so ds_load latency hides
// under the WMMA pipe (partial s_wait_dscnt instead of full drains).
template<int NSUB, bool AF32, bool RELU, bool OF32>
__global__ __launch_bounds__(256) void gemm_kernel(
    const void* __restrict__ Ap, const __bf16* __restrict__ Wt,
    const float* __restrict__ bias, void* __restrict__ Cp,
    int M, int N, int K)
{
  __shared__ __bf16 sA[128 * 32];          // A tile, row-major, stride 32
  __shared__ __bf16 sB[16 * NSUB * 32];    // Wt tile: (16*NSUB) cols x 32 K, stride 32

  const int t    = threadIdx.x;
  const int wave = t >> 5;
  const int lane = t & 31;
  const int blockM = blockIdx.x * 128;
  const int tileN0 = blockIdx.y * (16 * NSUB);

  v8f acc[NSUB];
  #pragma unroll
  for (int s = 0; s < NSUB; ++s) acc[s] = v8f{};

  const int arow = t >> 1, ahalf = t & 1;
  const int nn  = lane & 15;
  const int kbB = (lane >> 4) << 3;

  constexpr int GS = (NSUB < 4) ? NSUB : 4;   // fragment group size
  constexpr int NG = NSUB / GS;               // number of groups

  for (int k0 = 0; k0 < K; k0 += 32) {
    // ---- stage A: 128x32, 16 elements per thread ----
    {
      int di = arow * 32 + ahalf * 16;
      if constexpr (AF32) {
        const float* srcf = (const float*)Ap + (size_t)(blockM + arow) * K + k0 + ahalf * 16;
        if (k0 + 32 < K) __builtin_prefetch(srcf + 32, 0, 3);
        const float4* s4 = (const float4*)srcf;
        float4 f0 = s4[0], f1 = s4[1], f2 = s4[2], f3 = s4[3];
        sA[di +  0]=(__bf16)f0.x; sA[di +  1]=(__bf16)f0.y; sA[di +  2]=(__bf16)f0.z; sA[di +  3]=(__bf16)f0.w;
        sA[di +  4]=(__bf16)f1.x; sA[di +  5]=(__bf16)f1.y; sA[di +  6]=(__bf16)f1.z; sA[di +  7]=(__bf16)f1.w;
        sA[di +  8]=(__bf16)f2.x; sA[di +  9]=(__bf16)f2.y; sA[di + 10]=(__bf16)f2.z; sA[di + 11]=(__bf16)f2.w;
        sA[di + 12]=(__bf16)f3.x; sA[di + 13]=(__bf16)f3.y; sA[di + 14]=(__bf16)f3.z; sA[di + 15]=(__bf16)f3.w;
      } else {
        const __bf16* src = (const __bf16*)Ap + (size_t)(blockM + arow) * K + k0 + ahalf * 16;
        if (k0 + 32 < K) __builtin_prefetch(src + 32, 0, 3);
        *(uint4*)(&sA[di])     = *(const uint4*)(src);
        *(uint4*)(&sA[di + 8]) = *(const uint4*)(src + 8);
      }
    }
    // ---- stage B: (16*NSUB) x 32 in 16B chunks; fully unrolled, clause-friendly ----
    #pragma unroll
    for (int i = 0; i < (NSUB + 3) / 4; ++i) {
      int c = t + i * 256;
      if (64 * NSUB >= 256 || c < 64 * NSUB) {   // compile-time true for NSUB>=4
        int n = c >> 2, koff = (c & 3) * 8;
        *(uint4*)(&sB[n * 32 + koff]) =
            *(const uint4*)(&Wt[(size_t)(tileN0 + n) * K + k0 + koff]);
      }
    }
    __syncthreads();

    // ---- A fragment: 16-bit A 16x32 layout ----
    const int m   = lane & 15;
    const int kbA = (lane >> 4) << 3;
    const __bf16* ap = &sA[(wave * 16 + m) * 32 + kbA];
    v8bf alo = *(const v8bf*)ap;          // K = kbA .. kbA+7
    v8bf ahi = *(const v8bf*)(ap + 16);   // K = kbA+16 .. kbA+23
    v16bf a = __builtin_shufflevector(alo, ahi, 0,1,2,3,4,5,6,7,8,9,10,11,12,13,14,15);

    // ---- B fragments: double-buffered groups of GS ----
    auto loadB = [&](int s) -> v16bf {
      const __bf16* bp = &sB[(s * 16 + nn) * 32 + kbB];
      v8bf lo = *(const v8bf*)bp;
      v8bf hi = *(const v8bf*)(bp + 16);
      return __builtin_shufflevector(lo, hi, 0,1,2,3,4,5,6,7,8,9,10,11,12,13,14,15);
    };

    v16bf bf[2][GS];
    #pragma unroll
    for (int q = 0; q < GS; ++q) bf[0][q] = loadB(q);
    #pragma unroll
    for (int g = 0; g < NG; ++g) {
      const int cur = g & 1, nxt = cur ^ 1;
      if (g + 1 < NG) {
        #pragma unroll
        for (int q = 0; q < GS; ++q) bf[nxt][q] = loadB((g + 1) * GS + q);
      }
      #pragma unroll
      for (int q = 0; q < GS; ++q)
        acc[g * GS + q] = __builtin_amdgcn_wmma_f32_16x16x32_bf16(
            false, a, false, bf[cur][q], (short)0, acc[g * GS + q], false, false);
    }
    __syncthreads();
  }

  // ---- epilogue: lanes 0-15 -> M=v, lanes 16-31 -> M=8+v; N = lane&15 ----
  const int rbase = blockM + wave * 16 + ((lane >> 4) << 3);
  #pragma unroll
  for (int s = 0; s < NSUB; ++s) {
    const int col  = tileN0 + s * 16 + (lane & 15);
    const float bv = bias[col];
    #pragma unroll
    for (int v = 0; v < 8; ++v) {
      float o = acc[s][v] + bv;
      if (RELU) o = fmaxf(o, 0.0f);
      int row = rbase + v;
      if constexpr (OF32) ((float*)Cp)[(size_t)row * N + col] = o;
      else                ((__bf16*)Cp)[(size_t)row * N + col] = (__bf16)o;
    }
  }
}

// ---------------- layernorm over H=32, wave per row ----------------
__global__ __launch_bounds__(256) void ln_kernel(float* __restrict__ z,
    const float* __restrict__ scale, const float* __restrict__ bias)
{
  const int wave = threadIdx.x >> 5, lane = threadIdx.x & 31;
  const int row = blockIdx.x * 8 + wave;
  float v = z[(size_t)row * HL + lane];
  float s = v;
  #pragma unroll
  for (int o = 16; o > 0; o >>= 1) s += __shfl_xor(s, o, 32);
  float mu = s * (1.0f / 32.0f);
  float d = v - mu;
  float vs = d * d;
  #pragma unroll
  for (int o = 16; o > 0; o >>= 1) vs += __shfl_xor(vs, o, 32);
  float var = vs * (1.0f / 32.0f);
  z[(size_t)row * HL + lane] = d * rsqrtf(var + LN_EPS) * scale[lane] + bias[lane];
}

// ---------------- residual VQ: wave per row; codebooks bf16 in LDS, [lvl][dim][code] ----------------
__global__ __launch_bounds__(256) void vq_kernel(
    const float* __restrict__ z, const float* __restrict__ codebooks,
    float* __restrict__ out_idx, float* __restrict__ out_rep,
    __bf16* __restrict__ rep_bf, float* __restrict__ commit, int rowsPerWave)
{
  __shared__ __bf16 scb[NLVL * HL * NCODES];   // 48 KB, transposed: [lvl][h][code]
  __shared__ float  sres[8][HL];
  const int t = threadIdx.x, wave = t >> 5, lane = t & 31;

  for (int i = t; i < NLVL * HL * NCODES; i += 256) {
    int lvl = i / (HL * NCODES);
    int rem = i - lvl * (HL * NCODES);
    int h = rem / NCODES;
    int code = rem - h * NCODES;
    scb[i] = (__bf16)codebooks[((size_t)lvl * NCODES + code) * HL + h];
  }
  __syncthreads();

  float csum = 0.0f;
  for (int r = 0; r < rowsPerWave; ++r) {
    const int row = (blockIdx.x * 8 + wave) * rowsPerWave + r;
    sres[wave][lane] = z[(size_t)row * HL + lane];
    float rep_l = 0.0f;
    __syncthreads();
    #pragma unroll
    for (int lvl = 0; lvl < NLVL; ++lvl) {
      const int lbase = lvl * HL * NCODES;
      float best = 3.4e38f; int bi = 0;
      #pragma unroll
      for (int j = 0; j < 8; ++j) {
        int ci = lane * 8 + j;
        float d = 0.0f;
        #pragma unroll
        for (int h = 0; h < HL; ++h) {
          float diff = sres[wave][h] - (float)scb[lbase + h * NCODES + ci];
          d = fmaf(diff, diff, d);
        }
        if (d < best) { best = d; bi = ci; }
      }
      // wave-wide argmin allreduce (first-min tie-break like jnp.argmin)
      #pragma unroll
      for (int o = 16; o > 0; o >>= 1) {
        float ob = __shfl_xor(best, o, 32);
        int   oi = __shfl_xor(bi, o, 32);
        if (ob < best || (ob == best && oi < bi)) { best = ob; bi = oi; }
      }
      if (lane == 0) { out_idx[(size_t)row * 3 + lvl] = (float)bi; csum += best; }
      float q = codebooks[((size_t)lvl * NCODES + bi) * HL + lane];  // f32 gather for accuracy
      rep_l += q;
      __syncthreads();
      sres[wave][lane] -= q;
      __syncthreads();
    }
    out_rep[(size_t)row * HL + lane] = rep_l;
    rep_bf [(size_t)row * HL + lane] = (__bf16)rep_l;
  }
  if (lane == 0) atomicAdd(commit, csum * (BETA / ((float)B_ROWS * (float)HL)));
}

__global__ void init_kernel(float* commit) {
  if (threadIdx.x == 0 && blockIdx.x == 0) *commit = 0.0f;
}

// ---------------- host ----------------
extern "C" void kernel_launch(void* const* d_in, const int* in_sizes, int n_in,
                              void* d_out, int out_size, void* d_ws, size_t ws_size,
                              hipStream_t stream) {
  const float* ew[4] = {(const float*)d_in[0], (const float*)d_in[2],
                        (const float*)d_in[4], (const float*)d_in[6]};
  const float* eb[4] = {(const float*)d_in[1], (const float*)d_in[3],
                        (const float*)d_in[5], (const float*)d_in[7]};
  const float* dw[4] = {(const float*)d_in[8],  (const float*)d_in[10],
                        (const float*)d_in[12], (const float*)d_in[14]};
  const float* db[4] = {(const float*)d_in[9],  (const float*)d_in[11],
                        (const float*)d_in[13], (const float*)d_in[15]};
  const float* ln_scale  = (const float*)d_in[16];
  const float* ln_bias   = (const float*)d_in[17];
  const float* codebooks = (const float*)d_in[18];
  const float* x         = (const float*)d_in[19];

  // workspace carve-up
  char* ws = (char*)d_ws;
  size_t wo = 0;
  const int eK[4] = {768, 512, 256, 128}, eN[4] = {512, 256, 128, 32};
  const int dK[4] = {32, 128, 256, 512},  dN[4] = {128, 256, 512, 768};
  __bf16* ewt[4]; __bf16* dwt[4];
  for (int i = 0; i < 4; ++i) { ewt[i] = (__bf16*)(ws + wo); wo += (size_t)eK[i]*eN[i]*2; }
  for (int i = 0; i < 4; ++i) { dwt[i] = (__bf16*)(ws + wo); wo += (size_t)dK[i]*dN[i]*2; }
  wo = (wo + 255) & ~(size_t)255;
  float* zbuf = (float*)(ws + wo);  wo += (size_t)B_ROWS * HL * 4;
  __bf16* rep_bf = (__bf16*)(ws + wo); wo += (size_t)B_ROWS * HL * 2;
  __bf16* actA = (__bf16*)(ws + wo); wo += (size_t)B_ROWS * 512 * 2;
  __bf16* actB = (__bf16*)(ws + wo); wo += (size_t)B_ROWS * 512 * 2;

  float* out    = (float*)d_out;
  float* o_idx  = out;                          // [B,3]
  float* o_rep  = out + (size_t)B_ROWS * 3;     // [B,32]
  float* o_rec  = out + (size_t)B_ROWS * 35;    // [B,768]
  float* o_cmt  = out + (size_t)B_ROWS * 803;   // [1]

  // 1) weight convert+transpose
  for (int i = 0; i < 4; ++i) {
    int kn = eK[i] * eN[i];
    wconv_kernel<<<(kn + 255) / 256, 256, 0, stream>>>(ew[i], ewt[i], eK[i], eN[i]);
  }
  for (int i = 0; i < 4; ++i) {
    int kn = dK[i] * dN[i];
    wconv_kernel<<<(kn + 255) / 256, 256, 0, stream>>>(dw[i], dwt[i], dK[i], dN[i]);
  }

  dim3 blk(256);
  const int MB = B_ROWS / 128;
  // 2) encoder  (grid.y = N / (16*NSUB))
  gemm_kernel<16, true,  true,  false><<<dim3(MB, 2), blk, 0, stream>>>(x,    ewt[0], eb[0], actA, B_ROWS, 512, 768);
  gemm_kernel<16, false, true,  false><<<dim3(MB, 1), blk, 0, stream>>>(actA, ewt[1], eb[1], actB, B_ROWS, 256, 512);
  gemm_kernel< 8, false, true,  false><<<dim3(MB, 1), blk, 0, stream>>>(actB, ewt[2], eb[2], actA, B_ROWS, 128, 256);
  gemm_kernel< 2, false, false, true ><<<dim3(MB, 1), blk, 0, stream>>>(actA, ewt[3], eb[3], zbuf, B_ROWS,  32, 128);
  // 3) layernorm
  ln_kernel<<<B_ROWS / 8, blk, 0, stream>>>(zbuf, ln_scale, ln_bias);
  // 4) VQ
  init_kernel<<<1, 64, 0, stream>>>(o_cmt);
  vq_kernel<<<1024, blk, 0, stream>>>(zbuf, codebooks, o_idx, o_rep, rep_bf, o_cmt, 16);
  // 5) decoder
  gemm_kernel< 8, false, true,  false><<<dim3(MB, 1), blk, 0, stream>>>(rep_bf, dwt[0], db[0], actA, B_ROWS, 128,  32);
  gemm_kernel<16, false, true,  false><<<dim3(MB, 1), blk, 0, stream>>>(actA,   dwt[1], db[1], actB, B_ROWS, 256, 128);
  gemm_kernel<16, false, true,  false><<<dim3(MB, 2), blk, 0, stream>>>(actB,   dwt[2], db[2], actA, B_ROWS, 512, 256);
  gemm_kernel<16, false, false, true ><<<dim3(MB, 3), blk, 0, stream>>>(actA,   dwt[3], db[3], o_rec, B_ROWS, 768, 512);
}
